// GCN_local_MLP_13726715478779
// MI455X (gfx1250) — matrix operand order repaired and checked by reading, not compiled, not run on hardware
//
#include <hip/hip_runtime.h>
#include <stdint.h>

// ---------------------------------------------------------------------------
// Problem constants (B,N,M,H) = (200, 384, 512, 128)
// ---------------------------------------------------------------------------
constexpr int Bn = 200, Nn = 384, Mn = 512, Hn = 128;
constexpr int NN = Bn * Nn;        // 76800 nodes
constexpr int En = Bn * 2 * Mn;    // 204800 directed edges (excl. self loops)
constexpr int Rr = Bn * Mn;        // 102400 MLP rows
constexpr int ZW  = Mn + Nn + Mn;  // 1408  (z row width)
constexpr int ZCW = Mn + Nn + Nn;  // 1280  (zc row width)

typedef __bf16 v16bf __attribute__((ext_vector_type(16)));
typedef float  v8f   __attribute__((ext_vector_type(8)));

struct U8 { uint32_t v[8]; };
struct H8 { uint16_t h[8]; };

#define DEVFN __device__ __forceinline__

DEVFN uint16_t f2bf(float f) {
    uint32_t u = __builtin_bit_cast(uint32_t, f);
    u += 0x7FFFu + ((u >> 16) & 1u);            // round-to-nearest-even
    return (uint16_t)(u >> 16);
}
DEVFN float bf2f(uint16_t h) {
    uint32_t u = ((uint32_t)h) << 16;
    return __builtin_bit_cast(float, u);
}
DEVFN float wave_sum(float v) {
    v += __shfl_xor(v, 16, 32);
    v += __shfl_xor(v, 8, 32);
    v += __shfl_xor(v, 4, 32);
    v += __shfl_xor(v, 2, 32);
    v += __shfl_xor(v, 1, 32);
    return v;
}

// ISA 16-bit A layout (16x32): lane<16 -> K {0..7, 16..23}; lane>=16 -> K {8..15, 24..31}
DEVFN v16bf load_a_frag(const uint16_t* p, int hi) {
    const uint4 lo = *reinterpret_cast<const uint4*>(p + hi * 8);
    const uint4 hh = *reinterpret_cast<const uint4*>(p + 16 + hi * 8);
    U8 u{{lo.x, lo.y, lo.z, lo.w, hh.x, hh.y, hh.z, hh.w}};
    return __builtin_bit_cast(v16bf, u);
}
// ISA 16-bit B layout (32x16): lane<16 -> K {0..7, 16..23}; lane>=16 -> K {8..15, 24..31}
DEVFN v16bf load_b_frag(const uint16_t* p, int hi) {
    const uint4 a = *reinterpret_cast<const uint4*>(p + hi * 8);
    const uint4 b = *reinterpret_cast<const uint4*>(p + 16 + hi * 8);
    U8 u{{a.x, a.y, a.z, a.w, b.x, b.y, b.z, b.w}};
    return __builtin_bit_cast(v16bf, u);
}

// ---------------------------------------------------------------------------
// Degree / normalization
// ---------------------------------------------------------------------------
__global__ void k_init_deg(float* deg) {
    int i = blockIdx.x * 256 + threadIdx.x;
    deg[i] = 1.0f;                               // self loop
}
__global__ void k_deg_scatter(const int* __restrict__ dst, float* deg) {
    int e = blockIdx.x * 256 + threadIdx.x;
    unsafeAtomicAdd(&deg[dst[e]], 1.0f);
}
__global__ void k_rsqrt(float* deg) {
    int i = blockIdx.x * 256 + threadIdx.x;
    deg[i] = rsqrtf(deg[i]);
}

// ---------------------------------------------------------------------------
// GCN layer 1: aggregate the 2-dim input + norm-sum (linear layer commutes)
// ---------------------------------------------------------------------------
__global__ void k_aggx_init(const float* __restrict__ x, const float* __restrict__ dinv,
                            float* aggx, float* wsum) {
    int i = blockIdx.x * 256 + threadIdx.x;
    float w = dinv[i] * dinv[i];
    aggx[2 * i + 0] = w * x[2 * i + 0];
    aggx[2 * i + 1] = w * x[2 * i + 1];
    wsum[i] = w;
}
__global__ void k_aggx_scatter(const int* __restrict__ src, const int* __restrict__ dst,
                               const float* __restrict__ x, const float* __restrict__ dinv,
                               float* aggx, float* wsum) {
    int e = blockIdx.x * 256 + threadIdx.x;
    int s = src[e], d = dst[e];
    float nm = dinv[s] * dinv[d];
    unsafeAtomicAdd(&aggx[2 * d + 0], nm * x[2 * s + 0]);
    unsafeAtomicAdd(&aggx[2 * d + 1], nm * x[2 * s + 1]);
    unsafeAtomicAdd(&wsum[d], nm);
}
__global__ void k_h(const float* __restrict__ aggx, const float* __restrict__ wsum,
                    const float* __restrict__ W1, const float* __restrict__ b1,
                    float* __restrict__ h) {
    int i = blockIdx.x, j = threadIdx.x;
    float v = aggx[2 * i] * W1[j] + aggx[2 * i + 1] * W1[Hn + j] + wsum[i] * b1[j];
    h[(size_t)i * Hn + j] = v > 0.0f ? v : 0.0f;
}

// ---------------------------------------------------------------------------
// GCN layer 2: aggregate h (128-dim), then WMMA GEMM with W2
// ---------------------------------------------------------------------------
__global__ void k_aggH_init(const float* __restrict__ h, const float* __restrict__ dinv,
                            float* __restrict__ aggH) {
    int i = blockIdx.x, j = threadIdx.x;
    float w = dinv[i] * dinv[i];
    aggH[(size_t)i * Hn + j] = w * h[(size_t)i * Hn + j];
}
__global__ __launch_bounds__(256) void k_aggH_scatter(const int* __restrict__ src,
                                                      const int* __restrict__ dst,
                                                      const float* __restrict__ h,
                                                      const float* __restrict__ dinv,
                                                      float* aggH) {
    int e = (blockIdx.x * 256 + threadIdx.x) >> 5;   // one wave per edge
    int lane = threadIdx.x & 31;
    int s = src[e], d = dst[e];
    float nm = dinv[s] * dinv[d];
    int f = lane * 4;
    float4 hv = *reinterpret_cast<const float4*>(h + (size_t)s * Hn + f);
    float* o = aggH + (size_t)d * Hn + f;
    unsafeAtomicAdd(o + 0, nm * hv.x);
    unsafeAtomicAdd(o + 1, nm * hv.y);
    unsafeAtomicAdd(o + 2, nm * hv.z);
    unsafeAtomicAdd(o + 3, nm * hv.w);
}
__global__ void k_f32_to_bf16(const float* __restrict__ in, uint16_t* __restrict__ out) {
    size_t i = (size_t)blockIdx.x * 256 + threadIdx.x;
    out[i] = f2bf(in[i]);
}
// W (KxN row-major f32) -> Wt (NxK row-major bf16)
__global__ void k_transpose_bf16(const float* __restrict__ W, uint16_t* __restrict__ Wt,
                                 int K, int N) {
    int idx = blockIdx.x * 256 + threadIdx.x;
    int k = idx / N, n = idx - k * N;
    Wt[(size_t)n * K + k] = f2bf(W[idx]);
}

// ---------------------------------------------------------------------------
// Fused WMMA GEMM:  Out = relu(A @ Wt^T + bias * rowScale)  (bf16 in/out, f32 acc)
// block = 256 thr = 8 waves; wave tile = 16 rows x 64 cols; block tile = 128 x 64.
// Software-pipelined: ping-pong fragment sets so global loads for step k+32/k+64
// are in flight while the WMMAs for step k execute (K is a multiple of 64).
// ---------------------------------------------------------------------------
__global__ __launch_bounds__(256) void k_gemm_bf16(
    const uint16_t* __restrict__ A, const uint16_t* __restrict__ Wt,
    const float* __restrict__ bias, const float* __restrict__ rowScale,
    uint16_t* __restrict__ Out, int K, int Ncol) {
    const int w = threadIdx.x >> 5, lane = threadIdx.x & 31;
    const int rsel = lane & 15, hi = lane >> 4;
    const int row0 = blockIdx.x * 128 + w * 16;
    const int col0 = blockIdx.y * 64;

    const v8f zero = {0, 0, 0, 0, 0, 0, 0, 0};
    v8f acc[4];
#pragma unroll
    for (int t = 0; t < 4; ++t) acc[t] = zero;

    const uint16_t* arow = A + (size_t)(row0 + rsel) * K;
    const uint16_t* bcol[4];
#pragma unroll
    for (int t = 0; t < 4; ++t) bcol[t] = Wt + (size_t)(col0 + t * 16 + rsel) * K;

    // prologue: fragment set 0 for kk = 0
    v16bf a0 = load_a_frag(arow, hi);
    v16bf b0[4];
#pragma unroll
    for (int t = 0; t < 4; ++t) b0[t] = load_b_frag(bcol[t], hi);

    for (int kk = 0; kk < K; kk += 64) {
        // issue set-1 loads (kk+32) before consuming set 0
        v16bf a1 = load_a_frag(arow + kk + 32, hi);
        v16bf b1[4];
#pragma unroll
        for (int t = 0; t < 4; ++t) b1[t] = load_b_frag(bcol[t] + kk + 32, hi);
#pragma unroll
        for (int t = 0; t < 4; ++t)
            acc[t] = __builtin_amdgcn_wmma_f32_16x16x32_bf16(
                false, a0, false, b0[t], (short)0, acc[t], false, false);
        if (kk + 64 < K) {   // issue set-0 loads (kk+64) before consuming set 1
            a0 = load_a_frag(arow + kk + 64, hi);
#pragma unroll
            for (int t = 0; t < 4; ++t) b0[t] = load_b_frag(bcol[t] + kk + 64, hi);
        }
#pragma unroll
        for (int t = 0; t < 4; ++t)
            acc[t] = __builtin_amdgcn_wmma_f32_16x16x32_bf16(
                false, a1, false, b1[t], (short)0, acc[t], false, false);
    }
#pragma unroll
    for (int t = 0; t < 4; ++t) {
        const int col = col0 + t * 16 + rsel;
        const float bcolv = bias[col];
#pragma unroll
        for (int r = 0; r < 8; ++r) {
            const int row = row0 + r + hi * 8;
            const float sc = rowScale ? rowScale[row] : 1.0f;
            float v = acc[t][r] + bcolv * sc;
            v = v > 0.0f ? v : 0.0f;
            Out[(size_t)row * Ncol + col] = f2bf(v);
        }
    }
}

// ---------------------------------------------------------------------------
// Graph readout & feature assembly
// ---------------------------------------------------------------------------
__global__ void k_xg_sum(const uint16_t* __restrict__ xg16, uint16_t* __restrict__ xgg16) {
    int b = blockIdx.x, j = threadIdx.x;
    float s = 0.0f;
    for (int n = 0; n < Nn; ++n) s += bf2f(xg16[((size_t)b * Nn + n) * Hn + j]);
    xgg16[(size_t)b * Hn + j] = f2bf(s);
}
__global__ void k_feat(const int* __restrict__ ei0, const uint16_t* __restrict__ xg16,
                       const uint16_t* __restrict__ xgg16, uint16_t* __restrict__ feat) {
    int row = blockIdx.x;                // b*M + m
    int b = row / Mn, m = row - b * Mn;
    int t = threadIdx.x, seg = t >> 5, j = t & 31;
    const uint16_t* src;
    if (seg == 0)      src = xg16 + (size_t)ei0[(size_t)b * 2 * Mn + m] * Hn;
    else if (seg == 1) src = xg16 + (size_t)ei0[(size_t)b * 2 * Mn + Mn + m] * Hn;
    else               src = xgg16 + (size_t)b * Hn;
    const uint2 v = reinterpret_cast<const uint2*>(src)[j];
    reinterpret_cast<uint2*>(feat + (size_t)row * 384)[seg * 32 + j] = v;
}

// ---------------------------------------------------------------------------
// Narrow head GEMMs (N=4 / N=3): one wave per row; each lane reads contiguous
// 16-byte chunks (8 bf16) so a wave issues full 512 B coalesced requests.
// ---------------------------------------------------------------------------
__global__ __launch_bounds__(256) void k_head(const uint16_t* __restrict__ act,
                                              const float* __restrict__ W,
                                              const float* __restrict__ bias,
                                              float* __restrict__ out, int K, int O) {
    int row = (blockIdx.x * 256 + threadIdx.x) >> 5;
    int lane = threadIdx.x & 31;
    float s[4] = {0, 0, 0, 0};
    const uint16_t* a = act + (size_t)row * K;
    for (int base = lane * 8; base < K; base += 256) {
        uint4 pk = *reinterpret_cast<const uint4*>(a + base);
        H8 hh = __builtin_bit_cast(H8, pk);
#pragma unroll
        for (int j = 0; j < 8; ++j) {
            float av = bf2f(hh.h[j]);
            const float* wr = W + (size_t)(base + j) * O;
            for (int o = 0; o < O; ++o) s[o] += av * wr[o];
        }
    }
    for (int o = 0; o < O; ++o) s[o] = wave_sum(s[o]);
    if (lane == 0)
        for (int o = 0; o < O; ++o) out[(size_t)row * O + o] = s[o] + bias[o];
}

// ---------------------------------------------------------------------------
// Mask / sigmoid epilogue; writes p_fc and graph_topo sections of z
// ---------------------------------------------------------------------------
__global__ void k_epi1(const float* __restrict__ smlp, const float* __restrict__ cmlp,
                       const int* __restrict__ nsw, float* __restrict__ z,
                       float* __restrict__ vP, float* __restrict__ vC) {
    int idx = blockIdx.x * 256 + threadIdx.x;   // b*M + m
    int b = idx / Mn, m = idx - b * Mn;
    bool mask = m >= (Mn - nsw[b]);
    float s0 = smlp[(size_t)idx * 4 + 0], s1 = smlp[(size_t)idx * 4 + 1];
    float s2 = smlp[(size_t)idx * 4 + 2], s3 = smlp[(size_t)idx * 4 + 3];
    float c0 = cmlp[(size_t)idx * 3 + 0], c1 = cmlp[(size_t)idx * 3 + 1];
    float c2 = cmlp[(size_t)idx * 3 + 2];
    float topo = mask ? (1.0f / (1.0f + __expf(-s0))) : 1.0f;
    float pflow = mask ? s1 : c0;
    z[(size_t)b * ZW + m]            = pflow * topo;   // p_fc
    z[(size_t)b * ZW + Mn + Nn + m]  = topo;           // graph_topo
    vP[idx] = mask ? s2 : c1;
    vC[idx] = mask ? s3 : c2;
}

// ---------------------------------------------------------------------------
// v = inv_degree * (inc_parents@vP + inc_childs@vC); one wave per (b,n); 315 MB stream
// ---------------------------------------------------------------------------
__global__ __launch_bounds__(256) void k_v(const float* __restrict__ incP,
                                           const float* __restrict__ incC,
                                           const float* __restrict__ vP,
                                           const float* __restrict__ vC,
                                           const float* __restrict__ invdeg,
                                           float* __restrict__ vbuf, float* __restrict__ z) {
    int gw = (blockIdx.x * 256 + threadIdx.x) >> 5;   // b*N + n
    int lane = threadIdx.x & 31;
    int b = gw / Nn, n = gw - b * Nn;
    const float* rp = incP + (size_t)gw * Mn;
    const float* rc = incC + (size_t)gw * Mn;
    const float* wp = vP + (size_t)b * Mn;
    const float* wc = vC + (size_t)b * Mn;
    float s = 0.0f;
    for (int base = lane * 4; base < Mn; base += 128) {
        float4 a = *reinterpret_cast<const float4*>(rp + base);
        float4 p = *reinterpret_cast<const float4*>(wp + base);
        float4 c = *reinterpret_cast<const float4*>(rc + base);
        float4 q = *reinterpret_cast<const float4*>(wc + base);
        s += a.x * p.x + a.y * p.y + a.z * p.z + a.w * p.w;
        s += c.x * q.x + c.y * q.y + c.z * q.z + c.w * q.w;
    }
    s = wave_sum(s);
    if (lane == 0) {
        float val = (n == 0) ? 1.0f : invdeg[gw] * s;
        vbuf[gw] = val;
        z[(size_t)b * ZW + Mn + n] = val;              // v section of z
    }
}

// q_fc[b,m] = (sum_n A[n,m] * v[b,n]) * topo[b,m]
__global__ void k_qfc(const float* __restrict__ Amat, const float* __restrict__ vbuf,
                      const float* __restrict__ z, float* __restrict__ zc) {
    int b = blockIdx.x;
    int m = blockIdx.y * 256 + threadIdx.x;
    float s = 0.0f;
    for (int n = 0; n < Nn; ++n) s += Amat[(size_t)n * Mn + m] * vbuf[(size_t)b * Nn + n];
    float topo = z[(size_t)b * ZW + Mn + Nn + m];
    zc[(size_t)b * ZCW + m] = s * topo;
}

// pg/qg: one wave per (b,n), dual dot over A row (contiguous)
__global__ __launch_bounds__(256) void k_pgqg(const float* __restrict__ Amat,
                                              const float* __restrict__ x,
                                              const float* __restrict__ z,
                                              float* __restrict__ zc) {
    int gw = (blockIdx.x * 256 + threadIdx.x) >> 5;   // b*N + n
    int lane = threadIdx.x & 31;
    int b = gw / Nn, n = gw - b * Nn;
    const float* arow = Amat + (size_t)n * Mn;
    const float* pfc = z + (size_t)b * ZW;            // p_fc at z[..,0:M]
    const float* qfc = zc + (size_t)b * ZCW;          // q_fc at zc[..,0:M]
    float s1 = 0.0f, s2 = 0.0f;
    for (int base = lane * 4; base < Mn; base += 128) {
        float4 a = *reinterpret_cast<const float4*>(arow + base);
        float4 p = *reinterpret_cast<const float4*>(pfc + base);
        float4 q = *reinterpret_cast<const float4*>(qfc + base);
        s1 += a.x * p.x + a.y * p.y + a.z * p.z + a.w * p.w;
        s2 += a.x * q.x + a.y * q.y + a.z * q.z + a.w * q.w;
    }
    s1 = wave_sum(s1);
    s2 = wave_sum(s2);
    if (lane == 0) {
        float pl = x[(size_t)gw * 2 + 0];
        float ql = x[(size_t)gw * 2 + 1];
        zc[(size_t)b * ZCW + Mn + n]      = pl + s1;   // pg
        zc[(size_t)b * ZCW + Mn + Nn + n] = ql + s2;   // qg
    }
}

// ---------------------------------------------------------------------------
extern "C" void kernel_launch(void* const* d_in, const int* in_sizes, int n_in,
                              void* d_out, int out_size, void* d_ws, size_t ws_size,
                              hipStream_t stream) {
    (void)in_sizes; (void)n_in; (void)out_size; (void)ws_size;

    const float* x      = (const float*)d_in[0];
    const int*   ei0    = (const int*)d_in[1];
    const int*   ei1    = ei0 + En;
    const int*   nsw    = (const int*)d_in[2];
    const float* invdeg = (const float*)d_in[3];
    const float* incP   = (const float*)d_in[4];
    const float* incC   = (const float*)d_in[5];
    const float* Amat   = (const float*)d_in[6];
    const float* W1  = (const float*)d_in[7];
    const float* b1  = (const float*)d_in[8];
    const float* W2  = (const float*)d_in[9];
    const float* b2  = (const float*)d_in[10];
    const float* Ws1 = (const float*)d_in[11];
    const float* bs1 = (const float*)d_in[12];
    const float* Ws2 = (const float*)d_in[13];
    const float* bs2 = (const float*)d_in[14];
    const float* Wc1 = (const float*)d_in[15];
    const float* bc1 = (const float*)d_in[16];
    const float* Wc2 = (const float*)d_in[17];
    const float* bc2 = (const float*)d_in[18];

    float* z  = (float*)d_out;
    float* zc = z + (size_t)Bn * ZW;

    char* wsb = (char*)d_ws;
    size_t off = 0;
    auto take = [&](size_t bytes) -> void* {
        void* r = wsb + off;
        off = (off + bytes + 255) & ~(size_t)255;
        return r;
    };
    float*    deg    = (float*)take((size_t)NN * 4);          // deg -> dinv in place
    float*    aggx   = (float*)take((size_t)NN * 8);
    float*    wsum   = (float*)take((size_t)NN * 4);
    float*    h      = (float*)take((size_t)NN * Hn * 4);
    float*    aggH   = (float*)take((size_t)NN * Hn * 4);
    uint16_t* aggH16 = (uint16_t*)take((size_t)NN * Hn * 2);
    uint16_t* Wt2    = (uint16_t*)take((size_t)Hn * Hn * 2);
    uint16_t* xg16   = (uint16_t*)take((size_t)NN * Hn * 2);
    uint16_t* xgg16  = (uint16_t*)take((size_t)Bn * Hn * 2);
    uint16_t* feat   = (uint16_t*)take((size_t)Rr * 384 * 2);
    uint16_t* WtS1   = (uint16_t*)take((size_t)512 * 384 * 2);
    uint16_t* WtC1   = (uint16_t*)take((size_t)384 * 384 * 2);
    uint16_t* actS   = (uint16_t*)take((size_t)Rr * 512 * 2);
    uint16_t* actC   = (uint16_t*)take((size_t)Rr * 384 * 2);
    float*    smlp   = (float*)take((size_t)Rr * 4 * 4);
    float*    cmlp   = (float*)take((size_t)Rr * 3 * 4);
    float*    vP     = (float*)take((size_t)Rr * 4);
    float*    vC     = (float*)take((size_t)Rr * 4);
    float*    vbuf   = (float*)take((size_t)Bn * Nn * 4);

    // --- GCN normalization ---
    k_init_deg   <<<NN / 256, 256, 0, stream>>>(deg);
    k_deg_scatter<<<En / 256, 256, 0, stream>>>(ei1, deg);
    k_rsqrt      <<<NN / 256, 256, 0, stream>>>(deg);              // deg now holds dinv
    // --- layer 1 (linear commutes with segment-sum: aggregate 2-dim x + norm-sum) ---
    k_aggx_init   <<<NN / 256, 256, 0, stream>>>(x, deg, aggx, wsum);
    k_aggx_scatter<<<En / 256, 256, 0, stream>>>(ei0, ei1, x, deg, aggx, wsum);
    k_h           <<<NN, Hn, 0, stream>>>(aggx, wsum, W1, b1, h);
    // --- layer 2: aggregate h, then WMMA GEMM with W2 (bias scaled by norm-sum) ---
    k_aggH_init   <<<NN, Hn, 0, stream>>>(h, deg, aggH);
    k_aggH_scatter<<<En / 8, 256, 0, stream>>>(ei0, ei1, h, deg, aggH);
    k_f32_to_bf16 <<<(int)(((size_t)NN * Hn) / 256), 256, 0, stream>>>(aggH, aggH16);
    k_transpose_bf16<<<(Hn * Hn) / 256, 256, 0, stream>>>(W2, Wt2, Hn, Hn);
    k_gemm_bf16<<<dim3(NN / 128, Hn / 64), 256, 0, stream>>>(aggH16, Wt2, b2, wsum,
                                                             xg16, Hn, Hn);
    // --- readout + feature assembly ---
    k_xg_sum<<<Bn, Hn, 0, stream>>>(xg16, xgg16);
    k_feat  <<<Rr, 96, 0, stream>>>(ei0, xg16, xgg16, feat);
    // --- big MLPs (dominant ~70 GFLOP): bf16 WMMA with fused bias+relu ---
    k_transpose_bf16<<<(384 * 512) / 256, 256, 0, stream>>>(Ws1, WtS1, 384, 512);
    k_transpose_bf16<<<(384 * 384) / 256, 256, 0, stream>>>(Wc1, WtC1, 384, 384);
    k_gemm_bf16<<<dim3(Rr / 128, 512 / 64), 256, 0, stream>>>(feat, WtS1, bs1, nullptr,
                                                              actS, 384, 512);
    k_gemm_bf16<<<dim3(Rr / 128, 384 / 64), 256, 0, stream>>>(feat, WtC1, bc1, nullptr,
                                                              actC, 384, 384);
    // --- narrow heads ---
    k_head<<<Rr / 8, 256, 0, stream>>>(actS, Ws2, bs2, smlp, 512, 4);
    k_head<<<Rr / 8, 256, 0, stream>>>(actC, Wc2, bc2, cmlp, 384, 3);
    // --- epilogue / outputs ---
    k_epi1<<<Rr / 256, 256, 0, stream>>>(smlp, cmlp, nsw, z, vP, vC);
    k_v   <<<(Bn * Nn) / 8, 256, 0, stream>>>(incP, incC, vP, vC, invdeg, vbuf, z);
    k_qfc <<<dim3(Bn, Mn / 256), 256, 0, stream>>>(Amat, vbuf, z, zc);
    k_pgqg<<<(Bn * Nn) / 8, 256, 0, stream>>>(Amat, x, z, zc);
}
